// Model_90177133347650
// MI455X (gfx1250) — matrix operand order repaired and checked by reading, not compiled, not run on hardware
//
#include <hip/hip_runtime.h>
#include <hip/hip_bf16.h>

typedef __bf16 bf16;
typedef __attribute__((ext_vector_type(16))) __bf16 v16bf;
typedef __attribute__((ext_vector_type(8)))  __bf16 v8bf;
typedef __attribute__((ext_vector_type(8)))  float  v8f;

#define BM 128
#define BN 64
#define BK 32
#define LDK 40   // padded K stride (halves) -> 80B rows, 16B aligned

static __device__ __forceinline__ v16bf cat8(v8bf lo, v8bf hi) {
    return __builtin_shufflevector(lo, hi, 0,1,2,3,4,5,6,7,8,9,10,11,12,13,14,15);
}

// ---------------- cast fp32 -> bf16 ----------------
__global__ void cast_f32_bf16(const float* __restrict__ in, bf16* __restrict__ out, int n) {
    int i = blockIdx.x * blockDim.x + threadIdx.x;
    if (i < n) out[i] = (bf16)in[i];
}

// ---------------- transpose-cast: in [K,N] fp32 -> out [N,K] bf16 ----------------
__global__ void cast_t_f32_bf16(const float* __restrict__ in, bf16* __restrict__ out,
                                int K, int N)
{
    __shared__ float t[32][33];
    int kb = blockIdx.y * 32, nb = blockIdx.x * 32;
    int tx = threadIdx.x, ty = threadIdx.y;     // block (32,8)
    for (int i = ty; i < 32; i += 8) {
        int k = kb + i, n = nb + tx;
        t[i][tx] = (k < K && n < N) ? in[(size_t)k * N + n] : 0.f;
    }
    __syncthreads();
    for (int i = ty; i < 32; i += 8) {
        int n = nb + i, k = kb + tx;
        if (n < N && k < K) out[(size_t)n * K + k] = (bf16)t[tx][i];
    }
}

// ---------------- WMMA bf16 GEMM: C = act(A @ B^T + bias) ----------------
// A: [M,K] bf16 row-major. B: [N,K] bf16 row-major (weights pre-transposed).
// K % 32 == 0. Block tile 128x64, 8 waves x (16x64) = 4 WMMA accumulators each.
// Double-buffered LDS: one barrier per K-step; next-tile global loads overlap WMMA.
__global__ __launch_bounds__(256)
void gemm_bf16(const bf16* __restrict__ A, const bf16* __restrict__ B,
               const float* __restrict__ bias, float* __restrict__ Cf,
               bf16* __restrict__ Cb, int M, int N, int K, int relu)
{
    __shared__ __align__(16) bf16 As[2][BM][LDK];
    __shared__ __align__(16) bf16 Bs[2][BN][LDK];

    const int tid  = threadIdx.x;
    const int wave = tid >> 5;
    const int lane = tid & 31;
    const int bm = blockIdx.y * BM;
    const int bn = blockIdx.x * BN;
    const int wm = wave * 16;
    const int hl  = lane >> 4;
    const int l15 = lane & 15;

    v8f acc[4] = {{}, {}, {}, {}};

    const int aIdx = tid * 16;       // 4096 halves of A tile, 16/thread
    const int ar = aIdx >> 5;
    const int ac = aIdx & 31;        // 0 or 16
    const int bIdx = tid * 8;        // 2048 halves of B tile, 8/thread
    const int br = bIdx >> 5;
    const int bc = bIdx & 31;

    const int agr = bm + ar;
    const int bgn = bn + br;
    const bf16* pa = (agr < M) ? (A + (size_t)agr * K + ac) : nullptr;
    const bf16* pb = (bgn < N) ? (B + (size_t)bgn * K + bc) : nullptr;

    const int T = K / BK;

    // ---- stage tile 0 ----
    {
        v8bf a0 = {}, a1 = {}, b0 = {};
        if (pa) { a0 = ((const v8bf*)pa)[0]; a1 = ((const v8bf*)pa)[1]; }
        if (pb) { b0 = *(const v8bf*)pb; }
        *(v8bf*)&As[0][ar][ac]     = a0;
        *(v8bf*)&As[0][ar][ac + 8] = a1;
        *(v8bf*)&Bs[0][br][bc]     = b0;
    }
    __syncthreads();

    for (int t = 0; t < T; ++t) {
        const int cur = t & 1, nxt = cur ^ 1;
        const bool more = (t + 1) < T;

        // ---- issue next-tile global loads (fly under the WMMAs) ----
        v8bf na0 = {}, na1 = {}, nb0 = {};
        if (more) {
            const int ko = (t + 1) * BK;
            if (pa) {
                na0 = ((const v8bf*)(pa + ko))[0];
                na1 = ((const v8bf*)(pa + ko))[1];
                if (t + 2 < T) __builtin_prefetch(pa + ko + BK, 0, 1);
            }
            if (pb) {
                nb0 = *(const v8bf*)(pb + ko);
                if (t + 2 < T) __builtin_prefetch(pb + ko + BK, 0, 1);
            }
        }

        // ---- fragments per ISA 16-bit layouts, then 4 back-to-back WMMAs ----
        {
            const int arow = wm + l15;
            const int kh = hl * 8;
            v16bf a = cat8(*(const v8bf*)&As[cur][arow][kh],
                           *(const v8bf*)&As[cur][arow][kh + 16]);
            const int kb = hl * 8;
            v16bf bfr[4];
#pragma unroll
            for (int ct = 0; ct < 4; ++ct) {
                const int c = ct * 16 + l15;
                bfr[ct] = cat8(*(const v8bf*)&Bs[cur][c][kb],
                               *(const v8bf*)&Bs[cur][c][kb + 16]);
            }
#pragma unroll
            for (int ct = 0; ct < 4; ++ct)
                acc[ct] = __builtin_amdgcn_wmma_f32_16x16x32_bf16(
                              false, a, false, bfr[ct], (short)0, acc[ct], false, false);
        }

        // ---- store next tile into the other buffer; single barrier per step ----
        if (more) {
            *(v8bf*)&As[nxt][ar][ac]     = na0;
            *(v8bf*)&As[nxt][ar][ac + 8] = na1;
            *(v8bf*)&Bs[nxt][br][bc]     = nb0;
        }
        __syncthreads();
    }

    // ---- epilogue: C layout VGPR r -> (M = r + 8*hl, N = lane%16) ----
    const int row0 = bm + wm + hl * 8;
#pragma unroll
    for (int ct = 0; ct < 4; ++ct) {
        const int c = bn + ct * 16 + l15;
        if (c >= N) continue;
        float bv = bias ? bias[c] : 0.f;
#pragma unroll
        for (int r = 0; r < 8; ++r) {
            int gr = row0 + r;
            if (gr >= M) break;
            float v = acc[ct][r] + bv;
            if (relu) v = fmaxf(v, 0.f);
            size_t idx = (size_t)gr * N + c;
            if (Cf) Cf[idx] = v;
            if (Cb) Cb[idx] = (bf16)v;
        }
    }
}

// ---------------- per-node attention dot products ----------------
__global__ void node_dots(const float* __restrict__ xp, const float* __restrict__ a_src,
                          const float* __restrict__ a_dst, float* __restrict__ als,
                          float* __restrict__ ald, int H, int C)
{
    int n = blockIdx.x;
    int tid = threadIdx.x;
    __shared__ float red[128];
    for (int h = 0; h < H; ++h) {
        const float* row = xp + ((size_t)n * H + h) * C;
        float ps = 0.f, pd = 0.f;
        for (int c = tid; c < C; c += blockDim.x) {
            float v = row[c];
            ps += v * a_src[h * C + c];
            pd += v * a_dst[h * C + c];
        }
        red[tid] = ps; __syncthreads();
        for (int s = 64; s > 0; s >>= 1) { if (tid < s) red[tid] += red[tid + s]; __syncthreads(); }
        if (tid == 0) als[n * H + h] = red[0];
        __syncthreads();
        red[tid] = pd; __syncthreads();
        for (int s = 64; s > 0; s >>= 1) { if (tid < s) red[tid] += red[tid + s]; __syncthreads(); }
        if (tid == 0) ald[n * H + h] = red[0];
        __syncthreads();
    }
}

// order-preserving float<->uint for atomicMax
__device__ __forceinline__ unsigned fenc(float f) {
    unsigned u = __float_as_uint(f);
    return (u & 0x80000000u) ? ~u : (u | 0x80000000u);
}
__device__ __forceinline__ float fdec(unsigned u) {
    u = (u & 0x80000000u) ? (u & 0x7fffffffu) : ~u;
    return __uint_as_float(u);
}

__global__ void edge_logits(const int* __restrict__ src, const int* __restrict__ dst,
                            const float* __restrict__ als, const float* __restrict__ ald,
                            float* __restrict__ e, unsigned* __restrict__ mx, int E, int H)
{
    int i = blockIdx.x * blockDim.x + threadIdx.x;
    if (i >= E * H) return;
    int eid = i / H, h = i - eid * H;
    float v = als[src[eid] * H + h] + ald[dst[eid] * H + h];
    v = v > 0.f ? v : 0.2f * v;           // leaky_relu(0.2)
    e[i] = v;
    atomicMax(&mx[dst[eid] * H + h], fenc(v));
}

__global__ void edge_exp(const int* __restrict__ dst, const unsigned* __restrict__ mx,
                         float* __restrict__ e, float* __restrict__ denom, int E, int H)
{
    int i = blockIdx.x * blockDim.x + threadIdx.x;
    if (i >= E * H) return;
    int eid = i / H, h = i - eid * H;
    float m = fdec(mx[dst[eid] * H + h]);
    float ee = __expf(e[i] - m);
    e[i] = ee;
    atomicAdd(&denom[dst[eid] * H + h], ee);
}

__global__ void edge_scatter(const int* __restrict__ src, const int* __restrict__ dst,
                             const float* __restrict__ xp, const float* __restrict__ ee,
                             const float* __restrict__ denom, float* __restrict__ out,
                             int H, int C)
{
    int e = blockIdx.x;
    int s = src[e], d = dst[e];
    int HC = H * C;
    const float* xr = xp + (size_t)s * HC;
    float* orow = out + (size_t)d * HC;
    for (int idx = threadIdx.x; idx < HC; idx += blockDim.x) {
        int h = idx / C;
        float alpha = ee[(size_t)e * H + h] / (denom[d * H + h] + 1e-16f);
        atomicAdd(&orow[idx], xr[idx] * alpha);
    }
}

__global__ void bias_relu_cast(const float* __restrict__ in, const float* __restrict__ bias,
                               bf16* __restrict__ out, int total, int F)
{
    int i = blockIdx.x * blockDim.x + threadIdx.x;
    if (i >= total) return;
    int f = i % F;
    float v = in[i] + bias[f];
    out[i] = (bf16)fmaxf(v, 0.f);
}

// ---------------- host orchestration ----------------
struct Bump { char* base; size_t off; };
static void* bump(Bump& w, size_t bytes) {
    void* p = w.base + w.off;
    w.off = (w.off + bytes + 255) & ~(size_t)255;
    return p;
}
static dim3 cgrid(size_t n) { return dim3((unsigned)((n + 255) / 256)); }
static dim3 tgrid(int K, int N) { return dim3((unsigned)((N + 31) / 32), (unsigned)((K + 31) / 32)); }

static void run_side(const float* x, const int* ei, int E,
    const float* W1, const float* a1s, const float* a1d, const float* b1,
    const float* W2, const float* a2s, const float* a2d, const float* b2,
    const float* lw1, const float* lb1, const float* lw2, const float* lb2,
    const float* lw3, const float* lb3,
    int N, bf16* outK, Bump w, hipStream_t stream)
{
    const int F = 256, F2 = 512;
    bf16*  inbf = (bf16*)bump(w, (size_t)N * F * 2);
    bf16*  W1t  = (bf16*)bump(w, (size_t)F * F2 * 2);    // [F2,F]
    bf16*  W2t  = (bf16*)bump(w, (size_t)F2 * F * 2);    // [F,F2]
    bf16*  L1t  = (bf16*)bump(w, 256 * 256 * 2);         // [256,256]
    bf16*  L2t  = (bf16*)bump(w, 256 * 128 * 2);         // [128,256]
    bf16*  L3t  = (bf16*)bump(w, 128 * 64 * 2);          // [64,128]
    float* XP   = (float*)bump(w, (size_t)N * F2 * 4);
    float* AGG  = (float*)bump(w, (size_t)N * F2 * 4);
    bf16*  Hbf  = (bf16*)bump(w, (size_t)N * F2 * 2);
    bf16*  H2bf = (bf16*)bump(w, (size_t)N * 128 * 2);
    bf16*  Xbf  = (bf16*)bump(w, (size_t)N * F * 2);
    float* als  = (float*)bump(w, (size_t)N * 2 * 4);
    float* ald  = (float*)bump(w, (size_t)N * 2 * 4);
    unsigned* mx = (unsigned*)bump(w, (size_t)N * 2 * 4);
    float* denom = (float*)bump(w, (size_t)N * 2 * 4);
    float* ebuf  = (float*)bump(w, (size_t)E * 2 * 4);

    const int* src = ei;
    const int* dst = ei + E;
    dim3 blk(256);
    dim3 tblk(32, 8);
    unsigned gy = (unsigned)((N + BM - 1) / BM);

    cast_f32_bf16<<<cgrid((size_t)N * F), 256, 0, stream>>>(x, inbf, N * F);
    cast_t_f32_bf16<<<tgrid(F, F2),  tblk, 0, stream>>>(W1,  W1t, F,  F2);
    cast_t_f32_bf16<<<tgrid(F2, F),  tblk, 0, stream>>>(W2,  W2t, F2, F);
    cast_t_f32_bf16<<<tgrid(256, 256), tblk, 0, stream>>>(lw1, L1t, 256, 256);
    cast_t_f32_bf16<<<tgrid(256, 128), tblk, 0, stream>>>(lw2, L2t, 256, 128);
    cast_t_f32_bf16<<<tgrid(128, 64),  tblk, 0, stream>>>(lw3, L3t, 128, 64);

    // ---- GAT layer 1 (H=2, C=256) ----
    gemm_bf16<<<dim3(F2 / BN, gy), blk, 0, stream>>>(inbf, W1t, nullptr, XP, nullptr, N, F2, F, 0);
    node_dots<<<N, 128, 0, stream>>>(XP, a1s, a1d, als, ald, 2, F);
    hipMemsetAsync(mx, 0, (size_t)N * 2 * 4, stream);
    hipMemsetAsync(denom, 0, (size_t)N * 2 * 4, stream);
    hipMemsetAsync(AGG, 0, (size_t)N * F2 * 4, stream);
    edge_logits<<<cgrid((size_t)E * 2), 256, 0, stream>>>(src, dst, als, ald, ebuf, mx, E, 2);
    edge_exp<<<cgrid((size_t)E * 2), 256, 0, stream>>>(dst, mx, ebuf, denom, E, 2);
    edge_scatter<<<E, 256, 0, stream>>>(src, dst, XP, ebuf, denom, AGG, 2, F);
    bias_relu_cast<<<cgrid((size_t)N * F2), 256, 0, stream>>>(AGG, b1, Hbf, N * F2, F2);

    // ---- GAT layer 2 (H=1, C=256) ----
    gemm_bf16<<<dim3(F / BN, gy), blk, 0, stream>>>(Hbf, W2t, nullptr, XP, nullptr, N, F, F2, 0);
    node_dots<<<N, 128, 0, stream>>>(XP, a2s, a2d, als, ald, 1, F);
    hipMemsetAsync(mx, 0, (size_t)N * 4, stream);
    hipMemsetAsync(denom, 0, (size_t)N * 4, stream);
    hipMemsetAsync(AGG, 0, (size_t)N * F * 4, stream);
    edge_logits<<<cgrid((size_t)E), 256, 0, stream>>>(src, dst, als, ald, ebuf, mx, E, 1);
    edge_exp<<<cgrid((size_t)E), 256, 0, stream>>>(dst, mx, ebuf, denom, E, 1);
    edge_scatter<<<E, 256, 0, stream>>>(src, dst, XP, ebuf, denom, AGG, 1, F);
    bias_relu_cast<<<cgrid((size_t)N * F), 256, 0, stream>>>(AGG, b2, Xbf, N * F, F);

    // ---- MLP head 256 -> 256 -> 128 -> 64 (relu each) ----
    gemm_bf16<<<dim3(256 / BN, gy), blk, 0, stream>>>(Xbf,  L1t, lb1, nullptr, Hbf,  N, 256, 256, 1);
    gemm_bf16<<<dim3(128 / BN, gy), blk, 0, stream>>>(Hbf,  L2t, lb2, nullptr, H2bf, N, 128, 256, 1);
    gemm_bf16<<<dim3(64  / BN, gy), blk, 0, stream>>>(H2bf, L3t, lb3, nullptr, outK, N, 64, 128, 1);
}

extern "C" void kernel_launch(void* const* d_in, const int* in_sizes, int n_in,
                              void* d_out, int out_size, void* d_ws, size_t ws_size,
                              hipStream_t stream)
{
    const int F = 256, KOUT = 64;
    const float* x_m = (const float*)d_in[0];
    const float* x_d = (const float*)d_in[1];
    const int* eix = (const int*)d_in[2];
    const int* eiy = (const int*)d_in[3];
    int Mn = in_sizes[0] / F;
    int Dn = in_sizes[1] / F;
    int EX = in_sizes[2] / 2;
    int EY = in_sizes[3] / 2;

    Bump w{(char*)d_ws, 0};
    bf16* XK = (bf16*)bump(w, (size_t)Mn * KOUT * 2);
    bf16* YK = (bf16*)bump(w, (size_t)Dn * KOUT * 2);
    size_t mark = w.off;

    run_side(x_m, eix, EX,
             (const float*)d_in[4],  (const float*)d_in[5],  (const float*)d_in[6],  (const float*)d_in[7],
             (const float*)d_in[8],  (const float*)d_in[9],  (const float*)d_in[10], (const float*)d_in[11],
             (const float*)d_in[20], (const float*)d_in[21], (const float*)d_in[22], (const float*)d_in[23],
             (const float*)d_in[24], (const float*)d_in[25],
             Mn, XK, w, stream);

    w.off = mark;
    run_side(x_d, eiy, EY,
             (const float*)d_in[12], (const float*)d_in[13], (const float*)d_in[14], (const float*)d_in[15],
             (const float*)d_in[16], (const float*)d_in[17], (const float*)d_in[18], (const float*)d_in[19],
             (const float*)d_in[26], (const float*)d_in[27], (const float*)d_in[28], (const float*)d_in[29],
             (const float*)d_in[30], (const float*)d_in[31],
             Dn, YK, w, stream);

    // final: out[M,D] = XK @ YK^T  (bf16 in, fp32 out) -- YK is already [D,K] row-major
    dim3 blk(256);
    dim3 grd((unsigned)((Dn + BN - 1) / BN), (unsigned)((Mn + BM - 1) / BM));
    gemm_bf16<<<grd, blk, 0, stream>>>(XK, YK, nullptr, (float*)d_out, nullptr, Mn, Dn, KOUT, 0);
}
